// PIGNN_85555748537203
// MI455X (gfx1250) — matrix operand order repaired and checked by reading, not compiled, not run on hardware
//
#include <hip/hip_runtime.h>
#include <stdint.h>

// ---------------------------------------------------------------------------
// PIGNN forward for MI455X (gfx1250): bf16 WMMA GEMMs (f32 accumulate),
// async global->LDS staging (gfx1250 GLOBAL_LOAD_ASYNC_TO_LDS_B128),
// fused gather->GEMM->atomic-scatter msg phase, fused GEMM->residual update.
// Row-tiled blocks (R x 16 rows) to amortize B-fragment L2 traffic.
// ---------------------------------------------------------------------------

typedef __attribute__((ext_vector_type(16))) __bf16 v16bf;
typedef __attribute__((ext_vector_type(8)))  float  v8f;

#define N_NODES 100000
#define N_EDGES 600000
#define HDIM    128

__device__ __forceinline__ uint16_t f2bf(float f) {
    union { float f; uint32_t u; } x; x.f = f;
    uint32_t u = x.u;
    return (uint16_t)((u + 0x7FFFu + ((u >> 16) & 1u)) >> 16);   // RNE
}

// ---------------------------------------------------------------------------
// Async 16B global->LDS copy (ASYNCcnt-tracked).  The LDS byte address is the
// low 32 bits of the generic pointer (flat shared ptr = {aperture, lds_off}).
// ---------------------------------------------------------------------------
__device__ __forceinline__ void async_copy_b128(void* lds_dst, const void* gsrc) {
    uint32_t lds = (uint32_t)(uintptr_t)lds_dst;
    asm volatile("global_load_async_to_lds_b128 %0, %1, off"
                 :: "v"(lds), "v"(gsrc)
                 : "memory");
}
__device__ __forceinline__ void async_wait0() {
    asm volatile("s_wait_asynccnt 0x0" ::: "memory");
}

// ---------------------------------------------------------------------------
// Wave-level GEMM: R 16x16 output tiles sharing one B fragment stream.
// A tile in LDS, 16R rows x KTOT bf16, row stride LDA = KTOT+8 (16B pad ->
// conflict-free b128 LDS loads).
// A frag (16-bit 16x32): lane(m=lane&15, half=lane>>4) holds
//   k = kc*32 + 16*half + [0..15]
// B frag (16-bit 32x16): lane holds column n=lane&15,
//   k = kc*32 + 16*half + [0..15]  (contiguous 32B in transposed weights)
// ---------------------------------------------------------------------------
template<int KTOT, int R>
__device__ __forceinline__ void wmma_tiles_gemm(const uint16_t* aTile,
                                                const uint16_t* Wcol,
                                                int m, int half, v8f* c) {
    constexpr int LDA = KTOT + 8;
#pragma unroll
    for (int kc = 0; kc < KTOT / 32; ++kc) {
        union { v16bf v; uint4 q[2]; } b;
        const uint16_t* br = Wcol + kc * 32 + half * 16;
        b.q[0] = *(const uint4*)(br);
        b.q[1] = *(const uint4*)(br + 8);
#pragma unroll
        for (int t = 0; t < R; ++t) {
            union { v16bf v; uint4 q[2]; } a;
            const uint16_t* ar = aTile + (t * 16 + m) * LDA + kc * 32 + half * 16;
            a.q[0] = *(const uint4*)(ar);
            a.q[1] = *(const uint4*)(ar + 8);
            c[t] = __builtin_amdgcn_wmma_f32_16x16x32_bf16(
                false, a.v, false, b.v, (short)0, c[t], false, false);
        }
    }
}

// ---------------------------------------------------------------------------
// Weight prep: fp32 (K,N) row-major  ->  bf16 (N,K)  (transposed)
// ---------------------------------------------------------------------------
__global__ void transposeW_kernel(const float* __restrict__ W,
                                  uint16_t* __restrict__ Wt, int K, int N) {
    int i = blockIdx.x * blockDim.x + threadIdx.x;
    if (i >= K * N) return;
    int n = i / K, k = i % K;
    Wt[i] = f2bf(W[(size_t)k * N + n]);
}

// ---------------------------------------------------------------------------
// Encoder first linear (tiny K=9/10): out_bf = relu(X @ W + b), scalar fp32.
// ---------------------------------------------------------------------------
template<int KIN>
__global__ __launch_bounds__(256) void enc_l1_kernel(
    const float* __restrict__ X, const float* __restrict__ W,
    const float* __restrict__ b, uint16_t* __restrict__ out, int M) {
    int i = blockIdx.x * blockDim.x + threadIdx.x;
    if (i >= M * HDIM) return;
    int row = i >> 7, col = i & 127;
    float acc = b[col];
    const float* xr = X + (size_t)row * KIN;
#pragma unroll
    for (int k = 0; k < KIN; ++k) acc += xr[k] * W[k * HDIM + col];
    out[i] = f2bf(fmaxf(acc, 0.f));
}

// ---------------------------------------------------------------------------
// Generic WMMA linear: O(M,NCOLS) = act(A_bf(M,KTOT) @ Wt(NCOLS,KTOT) + b)
// Block = 16R rows x NCOLS cols; one wave per 16-col tile, R row tiles.
// ---------------------------------------------------------------------------
template<int KTOT, int NCOLS, int R, bool RELU>
__global__ __launch_bounds__(32 * (NCOLS / 16)) void wmma_linear_kernel(
    const uint16_t* __restrict__ A, const uint16_t* __restrict__ Wt,
    const float* __restrict__ bias, float* __restrict__ outF,
    uint16_t* __restrict__ outB) {
    constexpr int LDA = KTOT + 8;
    constexpr int NTHREADS = 32 * (NCOLS / 16);
    constexpr int U4_PER_ROW = KTOT / 8;
    __shared__ __align__(16) uint16_t aTile[16 * R * LDA];

    const int tid  = threadIdx.x;
    const int row0 = blockIdx.x * 16 * R;

    // Stage A tile with async global->LDS b128 copies.
    for (int i = tid; i < 16 * R * U4_PER_ROW; i += NTHREADS) {
        int r = i / U4_PER_ROW, cc = i % U4_PER_ROW;
        async_copy_b128(&aTile[r * LDA + cc * 8],
                        A + (size_t)(row0 + r) * KTOT + cc * 8);
    }
    async_wait0();
    __syncthreads();

    const int wave = tid >> 5;
    const int lane = tid & 31;
    const int half = lane >> 4;
    const int m    = lane & 15;
    const int col  = wave * 16 + m;

    v8f c[R] = {};
    wmma_tiles_gemm<KTOT, R>(aTile, Wt + (size_t)col * KTOT, m, half, c);

    const float bv = bias[col];
#pragma unroll
    for (int t = 0; t < R; ++t) {
#pragma unroll
        for (int r = 0; r < 8; ++r) {
            float v = c[t][r] + bv;
            if (RELU) v = fmaxf(v, 0.f);
            size_t g = (size_t)(row0 + t * 16 + half * 8 + r) * NCOLS + col;
            if (outF) outF[g] = v;
            if (outB) outB[g] = f2bf(v);
        }
    }
}

// ---------------------------------------------------------------------------
// Message phase: block of 64 edges; async-gather [h[src], h[dst], e] -> LDS
// (64x384 bf16), 48 WMMAs/wave, ReLU, atomic scatter-add into agg.
// ---------------------------------------------------------------------------
__global__ __launch_bounds__(256) void msg_kernel(
    const uint16_t* __restrict__ h_bf, const uint16_t* __restrict__ e_bf,
    const int* __restrict__ src, const int* __restrict__ dst,
    const uint16_t* __restrict__ Wt,   // (128, 384) bf16
    const float* __restrict__ bias,    // (128)
    float* __restrict__ agg) {
    constexpr int KTOT = 3 * HDIM, LDA = KTOT + 8, R = 4;
    __shared__ __align__(16) uint16_t aTile[16 * R * LDA];
    __shared__ int dstS[16 * R];

    const int tid = threadIdx.x;
    const int e0  = blockIdx.x * 16 * R;
    if (tid < 16 * R) dstS[tid] = dst[e0 + tid];

    // Gather: 64 rows x 3 segments x 16 b128 async copies.
    for (int i = tid; i < 16 * R * 48; i += 256) {
        int r = i / 48, j = i % 48, seg = j >> 4, off = j & 15;
        int e = e0 + r;
        const uint16_t* srow;
        if (seg == 0)      srow = h_bf + (size_t)src[e] * HDIM;
        else if (seg == 1) srow = h_bf + (size_t)dst[e] * HDIM;
        else               srow = e_bf + (size_t)e * HDIM;
        async_copy_b128(&aTile[r * LDA + seg * HDIM + off * 8], srow + off * 8);
    }
    async_wait0();
    __syncthreads();

    const int wave = tid >> 5;
    const int lane = tid & 31;
    const int half = lane >> 4;
    const int m    = lane & 15;
    const int col  = wave * 16 + m;

    v8f c[R] = {};
    wmma_tiles_gemm<KTOT, R>(aTile, Wt + (size_t)col * KTOT, m, half, c);

    const float bv = bias[col];
#pragma unroll
    for (int t = 0; t < R; ++t) {
#pragma unroll
        for (int r = 0; r < 8; ++r) {
            float v = fmaxf(c[t][r] + bv, 0.f);
            int node = dstS[t * 16 + half * 8 + r];
            atomicAdd(&agg[(size_t)node * HDIM + col], v);  // global_atomic_add_f32
        }
    }
}

// ---------------------------------------------------------------------------
// Update phase: A = [h_bf | bf16(agg)] (K=256); upd = relu(A@Wt+b);
// h += upd (fp32), refresh h_bf; zero agg in place for the next layer.
// Block = 32 nodes (R=2).
// ---------------------------------------------------------------------------
__global__ __launch_bounds__(256) void upd_kernel(
    float* __restrict__ h, uint16_t* __restrict__ h_bf,
    float* __restrict__ agg,
    const uint16_t* __restrict__ Wt,   // (128, 256) bf16
    const float* __restrict__ bias) {
    constexpr int KTOT = 2 * HDIM, LDA = KTOT + 8, R = 2;
    __shared__ __align__(16) uint16_t aTile[16 * R * LDA];

    const int tid  = threadIdx.x;
    const int row0 = blockIdx.x * 16 * R;

    // Stage h_bf (cols 0..127): 32 rows x 16 b128 async copies.
    for (int i = tid; i < 16 * R * 16; i += 256) {
        int r = i >> 4, off = i & 15;
        async_copy_b128(&aTile[r * LDA + off * 8],
                        h_bf + (size_t)(row0 + r) * HDIM + off * 8);
    }
    // Stage bf16(agg) (cols 128..255), zeroing agg after the read.
    for (int i = tid; i < 16 * R * 32; i += 256) {
        int r = i / 32, off = i % 32;                 // off in float4 units
        float* gp = agg + (size_t)(row0 + r) * HDIM + off * 4;
        float4 v = *(float4*)gp;
        *(float4*)gp = make_float4(0.f, 0.f, 0.f, 0.f);
        uint2 pk;
        pk.x = (uint32_t)f2bf(v.x) | ((uint32_t)f2bf(v.y) << 16);
        pk.y = (uint32_t)f2bf(v.z) | ((uint32_t)f2bf(v.w) << 16);
        *(uint2*)&aTile[r * LDA + HDIM + off * 4] = pk;
    }
    async_wait0();
    __syncthreads();

    const int wave = tid >> 5;
    const int lane = tid & 31;
    const int half = lane >> 4;
    const int m    = lane & 15;
    const int col  = wave * 16 + m;

    v8f c[R] = {};
    wmma_tiles_gemm<KTOT, R>(aTile, Wt + (size_t)col * KTOT, m, half, c);

    const float bv = bias[col];
#pragma unroll
    for (int t = 0; t < R; ++t) {
#pragma unroll
        for (int r = 0; r < 8; ++r) {
            float u = fmaxf(c[t][r] + bv, 0.f);
            size_t g = (size_t)(row0 + t * 16 + half * 8 + r) * HDIM + col;
            float hn = h[g] + u;
            h[g]    = hn;
            h_bf[g] = f2bf(hn);
        }
    }
}

// ---------------------------------------------------------------------------
// Decoder final layer (64 -> 15) + hard BC masks, scalar fp32.
// ---------------------------------------------------------------------------
__global__ __launch_bounds__(256) void dec_final_kernel(
    const float* __restrict__ F2,       // (N, 64) fp32
    const float* __restrict__ W3,       // (64, 15)
    const float* __restrict__ b3,       // (15)
    const float* __restrict__ bc_disp, const float* __restrict__ bc_rot,
    const float* __restrict__ face_mask,
    float* __restrict__ out, int M) {
    __shared__ float w3s[64 * 15];
    __shared__ float b3s[15];
    for (int i = threadIdx.x; i < 64 * 15; i += blockDim.x) w3s[i] = W3[i];
    for (int i = threadIdx.x; i < 15; i += blockDim.x) b3s[i] = b3[i];
    __syncthreads();

    int row = blockIdx.x * blockDim.x + threadIdx.x;
    if (row >= M) return;

    float f[64];
    const float4* fr = (const float4*)(F2 + (size_t)row * 64);
#pragma unroll
    for (int q = 0; q < 16; ++q) {
        float4 v = fr[q];
        f[q * 4 + 0] = v.x; f[q * 4 + 1] = v.y; f[q * 4 + 2] = v.z; f[q * 4 + 3] = v.w;
    }
    float dm = 1.f - bc_disp[row];
    float rm = 1.f - bc_rot[row];
    float fm0 = face_mask[row * 4 + 0], fm1 = face_mask[row * 4 + 1];
    float fm2 = face_mask[row * 4 + 2], fm3 = face_mask[row * 4 + 3];

#pragma unroll
    for (int o = 0; o < 15; ++o) {
        float acc = b3s[o];
#pragma unroll
        for (int k = 0; k < 64; ++k) acc += f[k] * w3s[k * 15 + o];
        float mk;
        if (o < 2)       mk = dm;
        else if (o == 2) mk = rm;
        else {
            int fi = (o - 3) / 3;
            mk = (fi == 0) ? fm0 : (fi == 1) ? fm1 : (fi == 2) ? fm2 : fm3;
        }
        out[(size_t)row * 15 + o] = acc * mk;
    }
}

// ---------------------------------------------------------------------------
// Host orchestration
// ---------------------------------------------------------------------------
extern "C" void kernel_launch(void* const* d_in, const int* in_sizes, int n_in,
                              void* d_out, int out_size, void* d_ws, size_t ws_size,
                              hipStream_t stream) {
    (void)in_sizes; (void)n_in; (void)out_size; (void)ws_size;
    const float* x         = (const float*)d_in[0];
    const float* edge_attr = (const float*)d_in[1];
    const int*   src       = (const int*)d_in[2];             // edge_index[0]
    const int*   dst       = ((const int*)d_in[2]) + N_EDGES; // edge_index[1]
    const float* bc_disp   = (const float*)d_in[3];
    const float* bc_rot    = (const float*)d_in[4];
    const float* face_mask = (const float*)d_in[5];
    const float* enc_n_W1  = (const float*)d_in[6];
    const float* enc_n_b1  = (const float*)d_in[7];
    const float* enc_n_W2  = (const float*)d_in[8];
    const float* enc_n_b2  = (const float*)d_in[9];
    const float* enc_e_W1  = (const float*)d_in[10];
    const float* enc_e_b1  = (const float*)d_in[11];
    const float* enc_e_W2  = (const float*)d_in[12];
    const float* enc_e_b2  = (const float*)d_in[13];
    const float* msg_W     = (const float*)d_in[14];   // (6, 384, 128)
    const float* msg_b     = (const float*)d_in[15];   // (6, 128)
    const float* upd_W     = (const float*)d_in[16];   // (6, 256, 128)
    const float* upd_b     = (const float*)d_in[17];   // (6, 128)
    const float* dec_W1    = (const float*)d_in[18];
    const float* dec_b1    = (const float*)d_in[19];
    const float* dec_W2    = (const float*)d_in[20];
    const float* dec_b2    = (const float*)d_in[21];
    const float* dec_W3    = (const float*)d_in[22];
    const float* dec_b3    = (const float*)d_in[23];
    float* out = (float*)d_out;

    // Workspace carve-up.
    uint8_t* ws = (uint8_t*)d_ws;
    size_t off = 0;
    auto take = [&](size_t bytes) -> uint8_t* {
        uint8_t* p = ws + off;
        off = (off + bytes + 255) & ~(size_t)255;
        return p;
    };
    uint16_t* wt_encn2 = (uint16_t*)take((size_t)128 * 128 * 2);
    uint16_t* wt_ence2 = (uint16_t*)take((size_t)128 * 128 * 2);
    uint16_t* wt_msg   = (uint16_t*)take((size_t)6 * 128 * 384 * 2);
    uint16_t* wt_upd   = (uint16_t*)take((size_t)6 * 128 * 256 * 2);
    uint16_t* wt_dec1  = (uint16_t*)take((size_t)128 * 128 * 2);
    uint16_t* wt_dec2  = (uint16_t*)take((size_t)64 * 128 * 2);
    float*    hF   = (float*)take((size_t)N_NODES * HDIM * 4);
    uint16_t* hB   = (uint16_t*)take((size_t)N_NODES * HDIM * 2);
    uint16_t* eB   = (uint16_t*)take((size_t)N_EDGES * HDIM * 2);
    float*    agg  = (float*)take((size_t)N_NODES * HDIM * 4);
    uint16_t* tmpB = (uint16_t*)take((size_t)N_EDGES * HDIM * 2);
    float*    f2F  = agg;   // reuse agg region for decoder hidden (N x 64 fp32)

    auto xpose = [&](const float* W, uint16_t* Wt, int K, int N) {
        int total = K * N;
        transposeW_kernel<<<(total + 255) / 256, 256, 0, stream>>>(W, Wt, K, N);
    };

    // 1) bf16-transpose all WMMA weights (~1 MB total, trivial).
    xpose(enc_n_W2, wt_encn2, 128, 128);
    xpose(enc_e_W2, wt_ence2, 128, 128);
    for (int l = 0; l < 6; ++l) {
        xpose(msg_W + (size_t)l * 384 * 128, wt_msg + (size_t)l * 128 * 384, 384, 128);
        xpose(upd_W + (size_t)l * 256 * 128, wt_upd + (size_t)l * 128 * 256, 256, 128);
    }
    xpose(dec_W1, wt_dec1, 128, 128);
    xpose(dec_W2, wt_dec2, 128, 64);

    // 2) zero the aggregation buffer once (upd_kernel re-zeros per layer).
    hipMemsetAsync(agg, 0, (size_t)N_NODES * HDIM * 4, stream);

    // 3) node encoder: relu(x@W1+b1) -> tmpB ; then WMMA 128x128 -> hF + hB.
    enc_l1_kernel<9><<<(N_NODES * HDIM + 255) / 256, 256, 0, stream>>>(
        x, enc_n_W1, enc_n_b1, tmpB, N_NODES);
    wmma_linear_kernel<128, 128, 2, false><<<N_NODES / 32, 256, 0, stream>>>(
        tmpB, wt_encn2, enc_n_b2, hF, hB);

    // 4) edge encoder: relu(ea@W1+b1) -> tmpB ; WMMA -> eB (bf16 only).
    enc_l1_kernel<10><<<(N_EDGES * HDIM + 255) / 256, 256, 0, stream>>>(
        edge_attr, enc_e_W1, enc_e_b1, tmpB, N_EDGES);
    wmma_linear_kernel<128, 128, 4, false><<<N_EDGES / 64, 256, 0, stream>>>(
        tmpB, wt_ence2, enc_e_b2, nullptr, eB);

    // 5) six message-passing layers.
    for (int l = 0; l < 6; ++l) {
        msg_kernel<<<N_EDGES / 64, 256, 0, stream>>>(
            hB, eB, src, dst,
            wt_msg + (size_t)l * 128 * 384, msg_b + (size_t)l * HDIM, agg);
        upd_kernel<<<N_NODES / 32, 256, 0, stream>>>(
            hF, hB, agg,
            wt_upd + (size_t)l * 128 * 256, upd_b + (size_t)l * HDIM);
    }

    // 6) decoder: H->H (relu) -> tmpB ; H->64 (relu) -> f2F ; 64->15 + masks.
    wmma_linear_kernel<128, 128, 2, true><<<N_NODES / 32, 256, 0, stream>>>(
        hB, wt_dec1, dec_b1, nullptr, tmpB);
    wmma_linear_kernel<128, 64, 2, true><<<N_NODES / 32, 128, 0, stream>>>(
        tmpB, wt_dec2, dec_b2, f2F, nullptr);
    dec_final_kernel<<<(N_NODES + 255) / 256, 256, 0, stream>>>(
        f2F, dec_W3, dec_b3, bc_disp, bc_rot, face_mask, out, N_NODES);
}
